// SelfAttention_47820165874100
// MI455X (gfx1250) — matrix operand order repaired and checked by reading, not compiled, not run on hardware
//
#include <hip/hip_runtime.h>

// MI455X / gfx1250, wave32. All matmuls run on v_wmma_f32_16x16x32_f16
// (f16 inputs, f32 accumulate). Problem: B=8,T=1024,C=1024,H=16,d=64.

typedef _Float16 f16;
typedef _Float16 v16h __attribute__((ext_vector_type(16)));
typedef _Float16 v8h  __attribute__((ext_vector_type(8)));
typedef float    v8f  __attribute__((ext_vector_type(8)));

#define CDIM   1024
#define TSEQ   1024
#define NBATCH 8
#define NHEAD  16
#define DHEAD  64
#define MROWS  (NBATCH * TSEQ)   // 8192

static __device__ __forceinline__ v8f wmma_f16(v16h a, v16h b, v8f c) {
  // 8 args: (neg_a, A, neg_b, B, c_mod, C, reuse_a, reuse_b)
  return __builtin_amdgcn_wmma_f32_16x16x32_f16(false, a, false, b, (short)0, c,
                                                false, false);
}

// ---- fragment loaders (ISA 7.12.2 layouts, wave32) ------------------------
// A 16x32 f16: lane L (<16): row=L, elems 0..7 = K k0..k0+7, 8..15 = k0+16..23
//              lane L (>=16): row=L-16, elems = K k0+8..15 / k0+24..31
static __device__ __forceinline__ v16h load_a_frag(const f16* base, int ld,
                                                   int row0, int k0, int lane) {
  const f16* p = base + (size_t)(row0 + (lane & 15)) * ld + k0 + ((lane & 16) ? 8 : 0);
  v8h lo = *(const v8h*)p;
  v8h hi = *(const v8h*)(p + 16);
  return __builtin_shufflevector(lo, hi, 0, 1, 2, 3, 4, 5, 6, 7,
                                 8, 9, 10, 11, 12, 13, 14, 15);
}

// B 32x16 f16 with B[k][n] = base[(col0+n)*ld + k0 + k]:
// lane L (<16): col=L, elems 0..7 = k0..k0+7, 8..15 = k0+16..23; lane>=16: col=L-16, k0+8..15 / k0+24..31
static __device__ __forceinline__ v16h load_b_frag(const f16* base, int ld,
                                                   int col0, int k0, int lane) {
  const f16* p = base + (size_t)(col0 + (lane & 15)) * ld + k0 + ((lane & 16) ? 8 : 0);
  v8h lo = *(const v8h*)p;
  v8h hi = *(const v8h*)(p + 16);
  return __builtin_shufflevector(lo, hi, 0, 1, 2, 3, 4, 5, 6, 7,
                                 8, 9, 10, 11, 12, 13, 14, 15);
}

// ---- conversion / transpose kernels ---------------------------------------
__global__ void cvt_f16_kernel(const float* __restrict__ in, f16* __restrict__ out, int n) {
  int i = blockIdx.x * blockDim.x + threadIdx.x;
  int stride = gridDim.x * blockDim.x;
  for (; i < n; i += stride) out[i] = (f16)in[i];
}

// W: [K=1024][N=1024] row-major f32  ->  Wt: [N][K] f16
__global__ void transpose_w_kernel(const float* __restrict__ W, f16* __restrict__ Wt) {
  int idx = blockIdx.x * blockDim.x + threadIdx.x;  // 1M elements
  int k = idx >> 10;
  int n = idx & 1023;
  Wt[(size_t)n * CDIM + k] = (f16)W[idx];
}

// ---- tiled WMMA GEMM: out[M=8192][N=1024] = A[M][K=1024] x Bt[N][K] + bias -
// Block: 8 waves as 2(M) x 4(N); wave tile 32x64 -> 8 independent accumulators.
// K loop fully unrolled: scheduler hoists load clauses across WMMAs.
// OUT_MODE 0: f16 row-major, 1: f16 V head-transposed [B,H,d,T], 2: f32 row-major
template <int OUT_MODE>
__global__ __launch_bounds__(256) void gemm_kernel(const f16* __restrict__ A,
                                                   const f16* __restrict__ Bt,
                                                   const float* __restrict__ bias,
                                                   void* __restrict__ out) {
  const int lane = threadIdx.x & 31;
  const int wave = threadIdx.x >> 5;
  const int wm = wave & 1, wn = wave >> 1;  // 2 x 4 wave grid
  const int m0 = blockIdx.y * 64 + wm * 32;
  const int n0 = blockIdx.x * 256 + wn * 64;

  v8f acc[2][4] = {};
#pragma unroll
  for (int kb = 0; kb < CDIM; kb += 32) {
    v16h af[2], bf[4];
#pragma unroll
    for (int i = 0; i < 2; ++i) af[i] = load_a_frag(A, CDIM, m0 + 16 * i, kb, lane);
#pragma unroll
    for (int j = 0; j < 4; ++j) bf[j] = load_b_frag(Bt, CDIM, n0 + 16 * j, kb, lane);
#pragma unroll
    for (int i = 0; i < 2; ++i)
#pragma unroll
      for (int j = 0; j < 4; ++j) acc[i][j] = wmma_f16(af[i], bf[j], acc[i][j]);
  }

  const int rbase = (lane & 16) ? 8 : 0;
#pragma unroll
  for (int i = 0; i < 2; ++i)
#pragma unroll
    for (int j = 0; j < 4; ++j) {
      int n = n0 + j * 16 + (lane & 15);
      float bv = bias[n];
#pragma unroll
      for (int r = 0; r < 8; ++r) {
        int m = m0 + i * 16 + rbase + r;
        float v = acc[i][j][r] + bv;
        if (OUT_MODE == 2) {
          ((float*)out)[(size_t)m * CDIM + n] = v;
        } else if (OUT_MODE == 0) {
          ((f16*)out)[(size_t)m * CDIM + n] = (f16)v;
        } else {  // V: [B,H,d,T]
          int h = n >> 6, dd = n & 63;
          int b = m >> 10, t = m & 1023;
          ((f16*)out)[((size_t)(b * NHEAD + h) * DHEAD + dd) * TSEQ + t] = (f16)v;
        }
      }
    }
}

// ---- fused flash attention (reference computes S = K . Q^T, causal) -------
// Grid: (B*H, T/128); 8 waves/block, each wave owns 16 rows of K (query role).
// Softmax is done in A-fragment (row-per-lane) layout after one LDS transpose:
// row reductions are in-lane + one shfl_xor(16); P lands directly in A layout.
__global__ __launch_bounds__(256) void attn_kernel(const f16* __restrict__ Qh,
                                                   const f16* __restrict__ Kh,
                                                   const f16* __restrict__ Vt,
                                                   f16* __restrict__ Oh) {
  __shared__ float sbuf[8][16][32];  // per-wave S transpose scratch (16 KB)
  const int lane = threadIdx.x & 31;
  const int wave = threadIdx.x >> 5;
  const int bh = blockIdx.x;  // b*NHEAD + h
  const int b = bh >> 4, h = bh & 15;
  const int i0 = blockIdx.y * 128 + wave * 16;

  const f16* Qb = Qh + (size_t)b * TSEQ * CDIM + h * DHEAD;
  const f16* Kb = Kh + (size_t)b * TSEQ * CDIM + h * DHEAD;
  const f16* Vb = Vt + (size_t)bh * DHEAD * TSEQ;

  // K rows act as the A matrix of the score WMMA; reused all steps.
  // Fold log2(e)/sqrt(C) into the fragment so scores come out in exp2 domain.
  v16h ka0 = load_a_frag(Kb, CDIM, i0, 0, lane);
  v16h ka1 = load_a_frag(Kb, CDIM, i0, 32, lane);
  const f16 kscale = (f16)0.045084223f;  // log2(e) * C^-0.5 = 1.4427/32
#pragma unroll
  for (int e = 0; e < 16; ++e) {
    ka0[e] *= kscale;
    ka1[e] *= kscale;
  }

  v8f o[4] = {};  // 16 x 64 unnormalized output accumulator (C layout)
  // per-lane row state: lane L (and L+16) owns row i0 + (L&15)
  float mrow = -__builtin_inff();
  float lrow = 0.f;

  const int rbase = (lane & 16) ? 8 : 0;  // C-layout row base
  const int colid = lane & 15;
  const int arow = lane & 15;             // A-layout row
  const int cbase = (lane & 16) ? 8 : 0;  // A-layout column base

  auto step = [&](int jb, bool masked) {
    // prefetch next step's Q rows and V columns into WGP$ (global_prefetch_b8)
    __builtin_prefetch(Qb + (size_t)(jb + 32 + colid) * CDIM, 0, 3);
    __builtin_prefetch(Vb + (size_t)colid * TSEQ + jb + 32, 0, 3);

    // scores for columns jb..jb+31: two 16-col fragments, K(d)=64 in two steps
    v16h qb0l = load_b_frag(Qb, CDIM, jb, 0, lane);
    v16h qb0h = load_b_frag(Qb, CDIM, jb, 32, lane);
    v16h qb1l = load_b_frag(Qb, CDIM, jb + 16, 0, lane);
    v16h qb1h = load_b_frag(Qb, CDIM, jb + 16, 32, lane);
    v8f s0 = {}, s1 = {};
    s0 = wmma_f16(ka0, qb0l, s0);
    s0 = wmma_f16(ka1, qb0h, s0);
    s1 = wmma_f16(ka0, qb1l, s1);
    s1 = wmma_f16(ka1, qb1h, s1);

    // transpose S via wave-private LDS (wave32 LDS is in-order; no barrier)
#pragma unroll
    for (int r = 0; r < 8; ++r) {
      sbuf[wave][rbase + r][colid] = s0[r];
      sbuf[wave][rbase + r][16 + colid] = s1[r];
    }
    // read back in A-fragment layout: lane holds row `arow`,
    // columns cbase..cbase+7 and cbase+16..cbase+23
    const float* sp = &sbuf[wave][arow][cbase];
    float sv[16];
#pragma unroll
    for (int e = 0; e < 8; ++e) {
      sv[e] = sp[e];
      sv[8 + e] = sp[16 + e];
    }

    if (masked) {  // causal: col > row -> -inf
#pragma unroll
      for (int e = 0; e < 16; ++e) {
        int col = jb + cbase + ((e < 8) ? e : 8 + (e - 8) + 8);  // e<8: +e, else +16+(e-8)
        if (col > i0 + arow) sv[e] = -__builtin_inff();
      }
    }

    // online softmax in exp2 domain; row reductions are in-lane + one shuffle
    float m16 = sv[0];
#pragma unroll
    for (int e = 1; e < 16; ++e) m16 = fmaxf(m16, sv[e]);
    m16 = fmaxf(m16, __shfl_xor(m16, 16, 32));
    float mn = fmaxf(mrow, m16);
    float fac = exp2f(mrow - mn);
    mrow = mn;

    float sum = 0.f;
    v16h pa;
#pragma unroll
    for (int e = 0; e < 16; ++e) {
      float p = exp2f(sv[e] - mn);
      sum += p;
      pa[e] = (f16)p;
    }
    sum += __shfl_xor(sum, 16, 32);
    lrow = lrow * fac + sum;

    // broadcast row rescale factors into C layout (vgpr r <-> row rbase+r)
    float facv[8];
#pragma unroll
    for (int r = 0; r < 8; ++r) facv[r] = __shfl(fac, rbase + r, 32);
#pragma unroll
    for (int t = 0; t < 4; ++t)
#pragma unroll
      for (int r = 0; r < 8; ++r) o[t][r] *= facv[r];

    // O += P x V  (V pre-transposed to [d][T] per head -> contiguous B frags)
#pragma unroll
    for (int t = 0; t < 4; ++t) {
      v16h vb = load_b_frag(Vb, TSEQ, t * 16, jb, lane);
      o[t] = wmma_f16(pa, vb, o[t]);
    }
  };

  int jb = 0;
  for (; jb + 31 <= i0; jb += 32) step(jb, false);  // full (unmasked) tiles
  for (; jb <= i0 + 15; jb += 32) step(jb, true);   // diagonal (masked) tiles

  // normalize (broadcast row sums into C layout) and store f16 head slice
  float lv[8];
#pragma unroll
  for (int r = 0; r < 8; ++r) lv[r] = 1.f / __shfl(lrow, rbase + r, 32);
#pragma unroll
  for (int t = 0; t < 4; ++t) {
#pragma unroll
    for (int r = 0; r < 8; ++r) {
      float v = o[t][r] * lv[r];
      int row = i0 + rbase + r;
      Oh[(size_t)(b * TSEQ + row) * CDIM + h * DHEAD + t * 16 + colid] = (f16)v;
    }
  }
}

extern "C" void kernel_launch(void* const* d_in, const int* in_sizes, int n_in,
                              void* d_out, int out_size, void* d_ws, size_t ws_size,
                              hipStream_t stream) {
  const float* x  = (const float*)d_in[0];
  const float* Wk = (const float*)d_in[1];
  const float* bk = (const float*)d_in[2];
  const float* Wq = (const float*)d_in[3];
  const float* bq = (const float*)d_in[4];
  const float* Wv = (const float*)d_in[5];
  const float* bv = (const float*)d_in[6];
  const float* Wp = (const float*)d_in[7];
  const float* bp = (const float*)d_in[8];

  char* ws = (char*)d_ws;
  const size_t MB = 1024 * 1024;
  // workspace plan (88 MB total):
  f16* xh  = (f16*)(ws + 0);        // 16 MB  x in f16
  f16* Qh  = (f16*)(ws + 16 * MB);  // 16 MB  Q [B*T, C]
  f16* Kh  = (f16*)(ws + 32 * MB);  // 16 MB  K [B*T, C]
  f16* Vt  = (f16*)(ws + 48 * MB);  // 16 MB  V [B,H,d,T]
  f16* Oh  = (f16*)(ws + 64 * MB);  // 16 MB  attn out [B*T, C]
  f16* Wkt = (f16*)(ws + 80 * MB);  // 2 MB each, transposed f16 weights
  f16* Wqt = (f16*)(ws + 82 * MB);
  f16* Wvt = (f16*)(ws + 84 * MB);
  f16* Wpt = (f16*)(ws + 86 * MB);

  hipLaunchKernelGGL(cvt_f16_kernel, dim3(2048), dim3(256), 0, stream, x, xh,
                     MROWS * CDIM);
  hipLaunchKernelGGL(transpose_w_kernel, dim3(4096), dim3(256), 0, stream, Wk, Wkt);
  hipLaunchKernelGGL(transpose_w_kernel, dim3(4096), dim3(256), 0, stream, Wq, Wqt);
  hipLaunchKernelGGL(transpose_w_kernel, dim3(4096), dim3(256), 0, stream, Wv, Wvt);
  hipLaunchKernelGGL(transpose_w_kernel, dim3(4096), dim3(256), 0, stream, Wp, Wpt);

  dim3 ggrid(CDIM / 256, MROWS / 64);
  hipLaunchKernelGGL((gemm_kernel<0>), ggrid, dim3(256), 0, stream, xh, Wqt, bq,
                     (void*)Qh);
  hipLaunchKernelGGL((gemm_kernel<0>), ggrid, dim3(256), 0, stream, xh, Wkt, bk,
                     (void*)Kh);
  hipLaunchKernelGGL((gemm_kernel<1>), ggrid, dim3(256), 0, stream, xh, Wvt, bv,
                     (void*)Vt);

  hipLaunchKernelGGL(attn_kernel, dim3(NBATCH * NHEAD, TSEQ / 128), dim3(256), 0,
                     stream, Qh, Kh, Vt, Oh);

  hipLaunchKernelGGL((gemm_kernel<2>), ggrid, dim3(256), 0, stream, Oh, Wpt, bp,
                     d_out);
}
